// VectorQuantizer_326417514521
// MI455X (gfx1250) — matrix operand order repaired and checked by reading, not compiled, not run on hardware
//
#include <hip/hip_runtime.h>
#include <hip/hip_bf16.h>

// ---------------------------------------------------------------------------
// VQ-VAE vector quantizer for MI455X (gfx1250, wave32).
//   inputs: [4,256,8,32,32] f32   -> N=32768 rows, C=256 (channel stride 8192)
//   emb:    [1024,256] f32        -> K=1024 codes
// GEMM x@emb^T done with split-precision bf16 WMMA (hi*hi + lo*hi + hi*lo),
// fused argmin, fused gather/write of quant (== quant_out), fused MSE loss.
// ---------------------------------------------------------------------------

typedef __attribute__((ext_vector_type(16))) __bf16 v16bf;
typedef __attribute__((ext_vector_type(8)))  float  v8f;

#define NROWS      32768
#define NCODES     1024
#define NCH        256
#define ROWTILES   2048          // NROWS / 16
#define CH_STRIDE  8192          // L*H*W
#define B_STRIDE   2097152       // C * CH_STRIDE
#define QUANT_ELEMS 8388608      // NROWS * NCH
#define FRAG_DW    8             // 8 dwords (16 bf16) per lane per fragment

__device__ __forceinline__ unsigned short f2bf(float x) {
  unsigned u = __float_as_uint(x);
  unsigned r = u + 0x7FFFu + ((u >> 16) & 1u);   // round-to-nearest-even
  return (unsigned short)(r >> 16);
}
__device__ __forceinline__ float bf2f(unsigned short b) {
  return __uint_as_float(((unsigned)b) << 16);
}
__device__ __forceinline__ v16bf pack2(uint4 a, uint4 b) {
  union { v16bf v; uint4 u[2]; } t;
  t.u[0] = a; t.u[1] = b;
  return t.v;
}

// ---------------------------------------------------------------------------
// Kernel 1: repack emb (f32) into hi/lo bf16 B-matrix fragments.
// Fragment f = kt*8 + cc covers codes [kt*16, kt*16+16) x channels [cc*32, cc*32+32).
// B-matrix 32x16 bf16 layout used here (contiguous k, not the ISA table order): lane L holds code N = L&15;
//   lanes 0-15:  K = cbase+0..15   (V0..V7, 2 bf16 per dword)
//   lanes 16-31: K = cbase+16..31
// -> lane's 16 bf16 are 16 *contiguous* channels of one emb row.
// ---------------------------------------------------------------------------
__global__ __launch_bounds__(256) void vq_prep(const float* __restrict__ emb,
                                               unsigned* __restrict__ embHi,
                                               unsigned* __restrict__ embLo) {
  int kt   = blockIdx.x;            // 0..63
  int cc   = threadIdx.x >> 5;      // 0..7  (wave id)
  int lane = threadIdx.x & 31;
  int code  = kt * 16 + (lane & 15);
  int cbase = cc * 32 + ((lane & 16) ? 16 : 0);
  const float* src = emb + (size_t)code * NCH + cbase;
  int f = kt * 8 + cc;
  unsigned* dh = embHi + (size_t)(f * 32 + lane) * FRAG_DW;
  unsigned* dl = embLo + (size_t)(f * 32 + lane) * FRAG_DW;
#pragma unroll
  for (int j = 0; j < 8; ++j) {
    float a = src[2 * j], b = src[2 * j + 1];
    unsigned short ah = f2bf(a), bh = f2bf(b);
    unsigned short al = f2bf(a - bf2f(ah)), bl = f2bf(b - bf2f(bh));
    dh[j] = (unsigned)ah | ((unsigned)bh << 16);
    dl[j] = (unsigned)al | ((unsigned)bl << 16);
  }
}

// ---------------------------------------------------------------------------
// Kernel 2: e2[k] = ||emb[k]||^2   (argmin key = e2 - 2*dot; x^2 is row-const)
// ---------------------------------------------------------------------------
__global__ __launch_bounds__(256) void vq_e2(const float* __restrict__ emb,
                                             float* __restrict__ e2) {
  int code = blockIdx.x * 16 + (threadIdx.x >> 4);   // 16 threads per code
  int seg  = (threadIdx.x & 15) * 16;
  const float* r = emb + (size_t)code * NCH + seg;
  float s = 0.f;
#pragma unroll
  for (int i = 0; i < 16; ++i) s += r[i] * r[i];
#pragma unroll
  for (int mask = 1; mask < 16; mask <<= 1) s += __shfl_xor(s, mask, 32);
  if ((threadIdx.x & 15) == 0) e2[code] = s;
}

// ---------------------------------------------------------------------------
// Kernel 3: main fused kernel. One workgroup (256 thr = 8 waves) per 16-row tile.
// ---------------------------------------------------------------------------
__global__ __launch_bounds__(256) void vq_main(const float* __restrict__ in,
                                               const float* __restrict__ emb,
                                               const unsigned* __restrict__ embHi,
                                               const unsigned* __restrict__ embLo,
                                               const float* __restrict__ e2,
                                               float* __restrict__ outQ,
                                               float* __restrict__ outIdx,
                                               float* __restrict__ partial) {
  __shared__ unsigned short xhi[16 * NCH];   // 8 KB  (row-major [m][c])
  __shared__ unsigned short xlo[16 * NCH];   // 8 KB
  __shared__ float          xf [16 * NCH];   // 16 KB
  __shared__ float wmin[8 * 16];
  __shared__ int   widx[8 * 16];
  __shared__ int   idxFinal[16];
  __shared__ float red[256];

  const int tid  = threadIdx.x;
  const int lane = tid & 31;
  const int wave = tid >> 5;

  const int n0 = blockIdx.x * 16;
  const int bI = n0 >> 13;                 // n0 / 8192
  const int r0 = n0 & 8191;
  const size_t base = (size_t)bI * B_STRIDE + r0;

  // ---- stage x tile: 16 rows x 256 channels; rows contiguous per channel ----
  {
    const int c = tid;  // one channel per thread
    const float4* g4 = (const float4*)(in + base + (size_t)c * CH_STRIDE);
#pragma unroll
    for (int q = 0; q < 4; ++q) {
      float4 v4 = g4[q];
      float vv[4] = {v4.x, v4.y, v4.z, v4.w};
#pragma unroll
      for (int i = 0; i < 4; ++i) {
        int m = q * 4 + i;
        float v = vv[i];
        unsigned short h = f2bf(v);
        unsigned short l = f2bf(v - bf2f(h));
        xhi[m * NCH + c] = h;
        xlo[m * NCH + c] = l;
        xf [m * NCH + c] = v;
      }
    }
  }
  __syncthreads();

  // ---- preload all A fragments (kt-invariant). A 16x32 bf16 layout used here (contiguous k, same as B):
  //  lanes 0-15 (M=lane):    V0..7 = K0..15
  //  lanes 16-31 (M=lane-16):V0..7 = K16..31
  const int M    = lane & 15;
  const int selA = (lane & 16) ? 16 : 0;
  v16bf Ah[8], Al[8];
#pragma unroll
  for (int cc = 0; cc < 8; ++cc) {
    int o1 = M * NCH + cc * 32 + selA;
    Ah[cc] = pack2(*(const uint4*)&xhi[o1], *(const uint4*)&xhi[o1 + 8]);
    Al[cc] = pack2(*(const uint4*)&xlo[o1], *(const uint4*)&xlo[o1 + 8]);
  }

  // ---- K loop: wave owns codes [wave*128, wave*128+128) = 8 k-tiles ----
  float minv[8];
  int   mini[8];
#pragma unroll
  for (int j = 0; j < 8; ++j) { minv[j] = 3.4e38f; mini[j] = 0; }

  for (int i = 0; i < 8; ++i) {
    const int kt = wave * 8 + i;
    v8f acc = {};
#pragma unroll
    for (int cc = 0; cc < 8; ++cc) {
      const int f = kt * 8 + cc;
      const uint4* bh = (const uint4*)(embHi + (size_t)(f * 32 + lane) * FRAG_DW);
      const uint4* bl = (const uint4*)(embLo + (size_t)(f * 32 + lane) * FRAG_DW);
      v16bf Bh = pack2(bh[0], bh[1]);
      v16bf Bl = pack2(bl[0], bl[1]);
      acc = __builtin_amdgcn_wmma_f32_16x16x32_bf16(false, Ah[cc], false, Bh,
                                                    (short)0, acc, false, false);
      acc = __builtin_amdgcn_wmma_f32_16x16x32_bf16(false, Al[cc], false, Bh,
                                                    (short)0, acc, false, false);
      acc = __builtin_amdgcn_wmma_f32_16x16x32_bf16(false, Ah[cc], false, Bl,
                                                    (short)0, acc, false, false);
    }
    // argmin update. C/D layout: VGPR j -> row M = j + 8*(lane>=16), N = lane&15
    const int code = kt * 16 + (lane & 15);
    const float e2v = e2[code];
#pragma unroll
    for (int j = 0; j < 8; ++j) {
      float d = e2v - 2.0f * acc[j];
      if (d < minv[j]) { minv[j] = d; mini[j] = code; }
    }
  }

  // ---- reduce argmin over the 16 lanes sharing a row-set ----
#pragma unroll
  for (int mask = 1; mask < 16; mask <<= 1) {
#pragma unroll
    for (int j = 0; j < 8; ++j) {
      float ov = __shfl_xor(minv[j], mask, 32);
      int   oi = __shfl_xor(mini[j], mask, 32);
      if (ov < minv[j] || (ov == minv[j] && oi < mini[j])) {
        minv[j] = ov; mini[j] = oi;
      }
    }
  }
  if ((lane & 15) == 0) {
    int rowBase = (lane & 16) ? 8 : 0;
#pragma unroll
    for (int j = 0; j < 8; ++j) {
      wmin[wave * 16 + rowBase + j] = minv[j];
      widx[wave * 16 + rowBase + j] = mini[j];
    }
  }
  __syncthreads();

  // ---- cross-wave argmin + idx output ----
  if (tid < 16) {
    float bv = wmin[tid];
    int   bi = widx[tid];
#pragma unroll
    for (int w = 1; w < 8; ++w) {
      float v = wmin[w * 16 + tid];
      int   i2 = widx[w * 16 + tid];
      if (v < bv || (v == bv && i2 < bi)) { bv = v; bi = i2; }
    }
    idxFinal[tid] = bi;
    outIdx[n0 + tid] = (float)bi;
  }
  __syncthreads();

  // ---- gather quant = emb[idx], write out (quant_out == quant), MSE partial ----
  float lsum = 0.f;
  {
    const int c = tid;
    float qv[16];
#pragma unroll
    for (int m = 0; m < 16; ++m) {
      int code = idxFinal[m];
      float q = emb[(size_t)code * NCH + c];
      qv[m] = q;
      float d = q - xf[m * NCH + c];
      lsum += d * d;
    }
    float* dst = outQ + base + (size_t)c * CH_STRIDE;
#pragma unroll
    for (int q4 = 0; q4 < 4; ++q4) {
      float4 v = make_float4(qv[4 * q4 + 0], qv[4 * q4 + 1],
                             qv[4 * q4 + 2], qv[4 * q4 + 3]);
      *(float4*)(dst + 4 * q4) = v;
    }
  }

  // deterministic block reduction of the squared-error partial
  red[tid] = lsum;
  __syncthreads();
#pragma unroll
  for (int o = 128; o > 0; o >>= 1) {
    if (tid < o) red[tid] += red[tid + o];
    __syncthreads();
  }
  if (tid == 0) partial[blockIdx.x] = red[0];
}

// ---------------------------------------------------------------------------
// Kernel 4: reduce 2048 partials -> loss = 1.25 * mse
// ---------------------------------------------------------------------------
__global__ __launch_bounds__(256) void vq_loss(const float* __restrict__ partial,
                                               float* __restrict__ outLoss) {
  __shared__ float red[256];
  float s = 0.f;
  for (int i = threadIdx.x; i < ROWTILES; i += 256) s += partial[i];
  red[threadIdx.x] = s;
  __syncthreads();
#pragma unroll
  for (int o = 128; o > 0; o >>= 1) {
    if (threadIdx.x < o) red[threadIdx.x] += red[threadIdx.x + o];
    __syncthreads();
  }
  if (threadIdx.x == 0)
    *outLoss = 1.25f * red[0] / (float)QUANT_ELEMS;
}

// ---------------------------------------------------------------------------
extern "C" void kernel_launch(void* const* d_in, const int* in_sizes, int n_in,
                              void* d_out, int out_size, void* d_ws, size_t ws_size,
                              hipStream_t stream) {
  (void)in_sizes; (void)n_in; (void)out_size; (void)ws_size;

  const float* in  = (const float*)d_in[0];   // [4,256,8,32,32] f32
  const float* emb = (const float*)d_in[1];   // [1024,256] f32

  float* out     = (float*)d_out;
  float* outQ    = out;                       // 8388608 elems
  float* outLoss = out + QUANT_ELEMS;         // 1 elem
  float* outIdx  = out + QUANT_ELEMS + 1;     // 32768 elems

  unsigned char* ws = (unsigned char*)d_ws;
  unsigned* embHi  = (unsigned*)(ws);                 // 512 KB
  unsigned* embLo  = (unsigned*)(ws + 524288);        // 512 KB
  float*    e2     = (float*)  (ws + 1048576);        // 4 KB
  float*    part   = (float*)  (ws + 1052672);        // 8 KB

  vq_prep<<<64,   256, 0, stream>>>(emb, embHi, embLo);
  vq_e2  <<<64,   256, 0, stream>>>(emb, e2);
  vq_main<<<ROWTILES, 256, 0, stream>>>(in, emb, embHi, embLo, e2,
                                        outQ, outIdx, part);
  vq_loss<<<1,    256, 0, stream>>>(part, outLoss);
}
